// DyCNN_76055280878097
// MI455X (gfx1250) — matrix operand order repaired and checked by reading, not compiled, not run on hardware
//
#include <hip/hip_runtime.h>

typedef __attribute__((ext_vector_type(16))) _Float16 v16h;
typedef __attribute__((ext_vector_type(8)))  _Float16 v8h;
typedef __attribute__((ext_vector_type(8)))  float    v8f;

#define BATCH 64
#define SEQ   512
#define DIN   256
#define NF    256
#define KS    3
#define KF    768            /* KS * NF */
#define LDA   264            /* slab row stride in halfs: 528 B, 16B aligned */
#define NCH1  (DIN / 32)     /* 8  K-chunks in stage 1 (over e)  */
#define NCH2  (KF / 32)      /* 24 K-chunks in stage 2 (over j)  */
#define NT16_V (KF / 16)     /* 48 column groups in Vsw  */
#define NT16_W (NF / 16)     /* 16 column groups in Wsw  */

// ---------------------------------------------------------------------------
// Swizzled B-operand storage: entry(chunk c, colgroup n16, lane) = 16 halfs,
// laid out so that lane L of a wave loads its WMMA B-fragment (32x16 f16,
// element j = B[16h + j][n], n = L&15, h = L>>4) as ONE contiguous 32-byte
// read:  base + ((c*NG + n16)*32 + L)*16 halfs.
// ---------------------------------------------------------------------------

// A-fragment from an LDS slab (16-bit A 16x32; k order chosen by this code, not the ISA table order):
// lane m = L&15, h = L>>4; elements 0..15 = K[kBase+16h .. +15]
// (contiguous, matching the B swizzle). Two aligned b128 DS loads.
__device__ __forceinline__ v16h frag_a(const _Float16* Xs, int rowBase, int kBase, int lane) {
  const int m = rowBase + (lane & 15);
  const int h = lane >> 4;
  const _Float16* p = Xs + m * LDA + kBase + 16 * h;
  union { v16h v; struct { v8h lo; v8h hi; } s; } u;
  u.s.lo = *(const v8h*)p;
  u.s.hi = *(const v8h*)(p + 8);
  return u.v;
}

// ---------------- prep: V (f32) -> Vsw (f16, fragment order) ----------------
__global__ __launch_bounds__(256)
void prep_V(const float* __restrict__ V, _Float16* __restrict__ Vsw) {
  const int idx = blockIdx.x * 256 + threadIdx.x;       // over 256*768 elements
  const int e  = idx / KF;
  const int kf = idx - e * KF;
  const int c1 = e >> 5, hh = (e >> 4) & 1, jj = e & 15;
  const int n16 = kf >> 4, n = kf & 15;
  const int lane = hh * 16 + n;
  Vsw[(((size_t)c1 * NT16_V + n16) * 32 + lane) * 16 + jj] = (_Float16)V[idx];
}

// ---------------- Stage 1: per-batch dynamic filter generation ----------------
// W[d, kf] = sum_e U[d,e]*sigmoid(z[b,e])*V[e,kf] + Bw[d,kf], written to Wsw in
// the stage-2 B-fragment order over (j = k*DIN+d, fo).
__global__ __launch_bounds__(256)
void dyconv_filter_gen(const float* __restrict__ U, const _Float16* __restrict__ Vsw,
                       const float* __restrict__ Bw, const float* __restrict__ z,
                       _Float16* __restrict__ Wsw) {
  __shared__ float    Zs[DIN];
  __shared__ _Float16 As[32 * LDA];
  const int b    = blockIdx.z;
  const int m0   = blockIdx.y * 32;    // d tile
  const int n0   = blockIdx.x * 256;   // kf tile
  const int t    = threadIdx.x;
  const int lane = t & 31;
  const int w    = t >> 5;
  const int wm   = w >> 2;             // 2 row groups of 16
  const int wn   = w & 3;              // 4 col groups of 64

  Zs[t] = 1.0f / (1.0f + __expf(-z[b * DIN + t]));
  __syncthreads();

  // Stage A slab once: (U ⊙ zf) rows m0..m0+31, e = 0..255 -> f16
#pragma unroll
  for (int i = 0; i < 8; ++i) {
    const int f4 = t + i * 256;        // 2048 float4 = 32 rows x 64
    const int r = f4 >> 6, c4 = f4 & 63;
    const float4 v = *(const float4*)&U[(size_t)(m0 + r) * DIN + 4 * c4];
    _Float16* dst = &As[r * LDA + 4 * c4];
    dst[0] = (_Float16)(v.x * Zs[4 * c4 + 0]);
    dst[1] = (_Float16)(v.y * Zs[4 * c4 + 1]);
    dst[2] = (_Float16)(v.z * Zs[4 * c4 + 2]);
    dst[3] = (_Float16)(v.w * Zs[4 * c4 + 3]);
  }
  __syncthreads();

  v8f acc[4] = {{}, {}, {}, {}};
  const int n16base = (n0 + wn * 64) >> 4;
  for (int c = 0; c < NCH1; ++c) {
    const v16h a = frag_a(As, wm * 16, c * 32, lane);
#pragma unroll
    for (int nb = 0; nb < 4; ++nb) {
      const v16h bf = *(const v16h*)(Vsw +
          (((size_t)c * NT16_V + n16base + nb) * 32 + lane) * 16);
      acc[nb] = __builtin_amdgcn_wmma_f32_16x16x32_f16(false, a, false, bf,
                                                       (short)0, acc[nb], false, false);
    }
  }

  // Epilogue: + B_w, pack 8 contiguous halfs per lane, store into Wsw fragment
  // order. j = k*DIN + d ; for this lane the 8 rows map to jj = 8h..8h+7.
  const int h = lane >> 4, n = lane & 15;
#pragma unroll
  for (int nb = 0; nb < 4; ++nb) {
    const int kf0   = n0 + wn * 64 + nb * 16;  // 16-col group (k constant inside)
    const int k     = kf0 >> 8;
    const int fo16  = (kf0 & 255) >> 4;
    const int kf    = kf0 + n;
    const int dbase = m0 + wm * 16 + 8 * h;    // 8 consecutive d rows
    v8h pk;
#pragma unroll
    for (int r = 0; r < 8; ++r)
      pk[r] = (_Float16)(acc[nb][r] + Bw[(size_t)(dbase + r) * KF + kf]);
    const int j     = k * DIN + dbase;
    const int cc    = j >> 5;
    const int lane2 = ((j >> 4) & 1) * 16 + n;
    _Float16* dst = Wsw + (((size_t)(b * NCH2 + cc) * NT16_W + fo16) * 32 + lane2) * 16 + 8 * h;
    *(v8h*)dst = pk;
  }
}

// ---------------- Stage 2: apply dynamic conv ----------------
// out[b,l,fo] = relu( sum_j patch[l,j] * W[j,fo] + bias[fo] ),
// patch[l, k*DIN+d] = x[b, l+k-1, d]. One x-slab (34 rows) covers all 3 taps.
__global__ __launch_bounds__(256)
void dyconv_apply(const float* __restrict__ x, const _Float16* __restrict__ Wsw,
                  const float* __restrict__ bias, float* __restrict__ out) {
  __shared__ _Float16 Xs[34 * LDA];
  const int b    = blockIdx.y;
  const int l0   = blockIdx.x * 32;
  const int t    = threadIdx.x;
  const int lane = t & 31;
  const int w    = t >> 5;
  const int wm   = w >> 2;             // 2 row groups of 16
  const int wn   = w & 3;              // 4 col groups of 64 -> full NF=256

  // Stage x slab: rows l0-1 .. l0+32 (zero-padded), cols 0..255 -> f16
  for (int f4 = t; f4 < 34 * 64; f4 += 256) {
    const int r = f4 >> 6, c4 = f4 & 63;
    const int gr = l0 + r - 1;
    float4 v = make_float4(0.f, 0.f, 0.f, 0.f);
    if (gr >= 0 && gr < SEQ)
      v = *(const float4*)&x[((size_t)b * SEQ + gr) * DIN + 4 * c4];
    _Float16* dst = &Xs[r * LDA + 4 * c4];
    dst[0] = (_Float16)v.x; dst[1] = (_Float16)v.y;
    dst[2] = (_Float16)v.z; dst[3] = (_Float16)v.w;
  }
  __syncthreads();

  const _Float16* Wb = Wsw + (size_t)b * NCH2 * NT16_W * 32 * 16;
  v8f acc[4] = {{}, {}, {}, {}};
  for (int c = 0; c < NCH2; ++c) {
    const int k  = c >> 3;             // tap
    const int d0 = (c & 7) * 32;       // channel chunk
    const v16h a = frag_a(Xs, wm * 16 + k, d0, lane);  // slab row = l_local + k
#pragma unroll
    for (int nb = 0; nb < 4; ++nb) {
      const int n16 = wn * 4 + nb;
      const v16h bf = *(const v16h*)(Wb + (((size_t)c * NT16_W + n16) * 32 + lane) * 16);
      acc[nb] = __builtin_amdgcn_wmma_f32_16x16x32_f16(false, a, false, bf,
                                                       (short)0, acc[nb], false, false);
    }
  }

  // Epilogue: bias + relu, f32 stores.
  const int h = lane >> 4, n = lane & 15;
#pragma unroll
  for (int nb = 0; nb < 4; ++nb) {
    const int fo = wn * 64 + nb * 16 + n;
    const float bv = bias[fo];
#pragma unroll
    for (int r = 0; r < 8; ++r) {
      const int l = l0 + wm * 16 + 8 * h + r;
      const float v = acc[nb][r] + bv;
      out[((size_t)b * SEQ + l) * NF + fo] = v > 0.f ? v : 0.f;
    }
  }
}

extern "C" void kernel_launch(void* const* d_in, const int* in_sizes, int n_in,
                              void* d_out, int out_size, void* d_ws, size_t ws_size,
                              hipStream_t stream) {
  (void)in_sizes; (void)n_in; (void)out_size; (void)ws_size;
  const float* x    = (const float*)d_in[0];
  const float* z    = (const float*)d_in[1];
  const float* U    = (const float*)d_in[2];
  const float* V    = (const float*)d_in[3];
  const float* Bw   = (const float*)d_in[4];
  const float* bias = (const float*)d_in[5];
  // d_in[6] is kernel_size (==3), baked in as KS.

  _Float16* Vsw = (_Float16*)d_ws;                          // 384 KiB
  _Float16* Wsw = (_Float16*)((char*)d_ws + (1u << 19));    // 24 MiB

  prep_V<<<dim3((DIN * KF) / 256), dim3(256), 0, stream>>>(V, Vsw);

  dim3 g1(KF / 256, DIN / 32, BATCH);   // (3, 8, 64)
  dyconv_filter_gen<<<g1, dim3(256), 0, stream>>>(U, Vsw, Bw, z, Wsw);

  dim3 g2(SEQ / 32, BATCH);             // (16, 64)
  dyconv_apply<<<g2, dim3(256), 0, stream>>>(x, Wsw, bias, (float*)d_out);
}
